// DeepSeekBlock_23476291240463
// MI455X (gfx1250) — matrix operand order repaired and checked by reading, not compiled, not run on hardware
//
#include <hip/hip_runtime.h>
#include <hip/hip_bf16.h>
#include <stdint.h>

typedef __bf16 bf16;
typedef __attribute__((ext_vector_type(16))) __bf16 v16bf;
typedef __attribute__((ext_vector_type(8)))  float  v8f;

union FragBF { uint4 q[2]; v16bf v; };

#define B_     2
#define S_     2048
#define D_     1024
#define T_     (B_*S_)
#define HEADS_ 16
#define HD_    64
#define LAT_   512
#define NE_    8
#define TOPK_  2
#define ED_    2048
#define MAXROWS_ 9216   /* 8192 + 8*128 pad, multiple of 128 */

__device__ __forceinline__ v8f wmma_bf16(const FragBF& a, const FragBF& b, v8f c) {
  return __builtin_amdgcn_wmma_f32_16x16x32_bf16(false, a.v, false, b.v, (short)0, c, false, false);
}

// CDNA5 async DMA: global -> LDS, 16B per lane, tracked by ASYNCcnt
__device__ __forceinline__ void async_b128(unsigned lds_byte_off, uint64_t gaddr) {
  asm volatile("global_load_async_to_lds_b128 %0, %1, off"
               :: "v"(lds_byte_off), "v"(gaddr) : "memory");
}
#define WAIT_ASYNC_HEX(h) asm volatile("s_wait_asynccnt " h ::: "memory")
__device__ __forceinline__ unsigned lds_off(const void* p) { return (unsigned)(uintptr_t)p; }

// ---------------------------------------------------------------- transpose+cvt: f32 [K,N] -> bf16 [N,K]
__global__ __launch_bounds__(256) void cvt_t_kernel(const float* __restrict__ W, bf16* __restrict__ Wt,
                                                    int K, int N) {
  __shared__ float tile[32][33];
  const int e = blockIdx.z;
  const float* Ws = W + (size_t)e*K*N;
  bf16* Wd = Wt + (size_t)e*K*N;
  const int n0 = blockIdx.x*32, k0 = blockIdx.y*32;
  const int tx = threadIdx.x, ty = threadIdx.y;
#pragma unroll
  for (int i=0;i<32;i+=8) tile[ty+i][tx] = Ws[(size_t)(k0+ty+i)*N + n0+tx];
  __syncthreads();
#pragma unroll
  for (int i=0;i<32;i+=8) Wd[(size_t)(n0+ty+i)*K + k0+tx] = (bf16)tile[tx][ty+i];
}

// ---------------------------------------------------------------- RMSNorm -> bf16
__global__ __launch_bounds__(256) void rmsnorm_kernel(const float* __restrict__ x, const float* __restrict__ gw,
                                                      bf16* __restrict__ out) {
  const int t = blockIdx.x, tid = threadIdx.x;
  const float* xr = x + (size_t)t*D_;
  float ss = 0.f;
  for (int i=tid;i<D_;i+=256) { float v = xr[i]; ss += v*v; }
#pragma unroll
  for (int off=16; off>=1; off>>=1) ss += __shfl_xor(ss, off);
  __shared__ float red[8];
  if ((tid&31)==0) red[tid>>5] = ss;
  __syncthreads();
  float tot = 0.f;
#pragma unroll
  for (int i=0;i<8;i++) tot += red[i];
  const float r = rsqrtf(tot/(float)D_ + 1e-6f);
  bf16* orow = out + (size_t)t*D_;
  for (int i=tid;i<D_;i+=256) orow[i] = (bf16)(xr[i]*r*gw[i]);
}

// ---------------------------------------------------------------- generic WMMA GEMM  C[M,N] = A[M,K] * Bt[N,K]^T
// Triple-buffered GLOBAL_LOAD_ASYNC_TO_LDS pipeline, one barrier per K=32 stage, branch-free steady state.
// mode 0: f32 out (+res)   1: bf16 out   2: bf16 transposed (b,col,s)   3: bf16 SiLU   4: f32 * rowgate -> out[slot[row]]
__global__ __launch_bounds__(256) void gemm_kernel(
    const bf16* __restrict__ A, const bf16* __restrict__ Bt,
    int M, int N, int K,
    const int* __restrict__ a_rowidx,
    const int* __restrict__ mblk_expert, long long b_expert_stride,
    int mode,
    float* __restrict__ outf, bf16* __restrict__ outb,
    const float* __restrict__ res, int seqS,
    const int* __restrict__ slot, const float* __restrict__ rowgate)
{
  __shared__ alignas(16) bf16 smem[3*2*128*40];   // 3 bufs x (A tile + B tile), 60KB
  const int tid = threadIdx.x;
  const int bm = blockIdx.y, bn = blockIdx.x;
  const int m0 = bm*128, n0 = bn*128;
  if (mblk_expert) Bt += (long long)mblk_expert[bm]*b_expert_stride;
  // loader mapping: thread covers half a row (16 elements = 2 x b128)
  const int row = tid>>1, sb = (tid&1)*2;
  int grow = a_rowidx ? a_rowidx[m0+row] : (m0+row);
  if (grow < 0) grow = 0;     // pad rows: any valid address; results discarded downstream
  const bf16* ga = A  + (size_t)grow*K      + sb*8;
  const bf16* gb = Bt + (size_t)(n0+row)*K  + sb*8;
  const unsigned lbase = lds_off(smem) + (unsigned)(row*40 + sb*8)*2;

  const int wave = tid>>5, lane = tid&31, g = lane>>4, ll = lane&15;
  const int wm = (wave&1)*64, wn = (wave>>1)*32;

  v8f acc[4][2];
#pragma unroll
  for (int i=0;i<4;i++)
#pragma unroll
    for (int j=0;j<2;j++)
#pragma unroll
      for (int r=0;r<8;r++) acc[i][j][r] = 0.f;

  int ibuf = 0, cbuf = 0;
  auto issue = [&]() {
    const unsigned la = lbase + (unsigned)ibuf*20480u;
    const unsigned lb = la + 10240u;
    const uint64_t a64 = (uint64_t)ga, b64 = (uint64_t)gb;
    async_b128(la,      a64);
    async_b128(la+16u,  a64+16u);
    async_b128(lb,      b64);
    async_b128(lb+16u,  b64+16u);
    ga += 32; gb += 32;
    ibuf = (ibuf==2) ? 0 : ibuf+1;
  };
  auto do_stage = [&]() {
    const bf16* cA = smem + cbuf*(2*128*40);
    const bf16* cB = cA + 128*40;
    FragBF af[4], bfr[2];
#pragma unroll
    for (int mi=0;mi<4;mi++) {
      const bf16* ab = cA + (wm + mi*16 + ll)*40;
      af[mi].q[0] = *(const uint4*)(ab + g*8);        // K = 8g..8g+7
      af[mi].q[1] = *(const uint4*)(ab + 16 + g*8);   // K = 16+8g..
    }
#pragma unroll
    for (int ni=0;ni<2;ni++) {
      const bf16* bb = cB + (wn + ni*16 + ll)*40 + g*8;  // K = 8g..8g+7, 16+8g..
      bfr[ni].q[0] = *(const uint4*)(bb);
      bfr[ni].q[1] = *(const uint4*)(bb + 16);
    }
#pragma unroll
    for (int mi=0;mi<4;mi++)
#pragma unroll
      for (int ni=0;ni<2;ni++)
        acc[mi][ni] = wmma_bf16(af[mi], bfr[ni], acc[mi][ni]);
    cbuf = (cbuf==2) ? 0 : cbuf+1;
  };

  const int nst = K >> 5;
  issue();
  for (int st=0; st<nst-1; ++st) {   // branch-free steady state
    issue();
    WAIT_ASYNC_HEX("0x4");
    __syncthreads();
    do_stage();
  }
  WAIT_ASYNC_HEX("0x0");             // peeled tail
  __syncthreads();
  do_stage();

#pragma unroll
  for (int mi=0;mi<4;mi++)
#pragma unroll
    for (int ni=0;ni<2;ni++)
#pragma unroll
      for (int r=0;r<8;r++) {
        const int rrow = m0 + wm + mi*16 + r + 8*g;
        const int col  = n0 + wn + ni*16 + ll;
        float v = acc[mi][ni][r];
        if (mode==0) {
          size_t idx = (size_t)rrow*N + col;
          outf[idx] = v + (res ? res[idx] : 0.f);
        } else if (mode==1) {
          outb[(size_t)rrow*N + col] = (bf16)v;
        } else if (mode==2) {
          int bb = rrow / seqS, s = rrow % seqS;
          outb[((size_t)bb*N + col)*seqS + s] = (bf16)v;
        } else if (mode==3) {
          float sg = v / (1.f + __expf(-v));
          outb[(size_t)rrow*N + col] = (bf16)sg;
        } else {
          int sl = slot[rrow];
          if (sl >= 0) outf[(size_t)sl*N + col] = rowgate[rrow]*v;
        }
      }
}

// ---------------------------------------------------------------- flash attention (causal), 1 wave per 16x64 Q tile
// Double-buffered async K/V staging; single-wave block => no barriers (DS ops are in-order per wave).
__global__ __launch_bounds__(32) void attn_kernel(
    const bf16* __restrict__ q, const bf16* __restrict__ k,
    const bf16* __restrict__ vt, bf16* __restrict__ o)
{
  __shared__ alignas(16) bf16 sk[2][32*72];   // 32 keys x 64 d (stride 72)
  __shared__ alignas(16) bf16 sv[2][64*40];   // 64 d x 32 keys (stride 40)
  __shared__ alignas(16) bf16 sp[16*40];      // P tile, 16 rows x 32 keys (stride 40)
  const int qt = blockIdx.x, h = blockIdx.y, b = blockIdx.z;
  const int q0 = qt*16;
  const int lane = threadIdx.x, g = lane>>4, ll = lane&15;

  auto issue_kv = [&](int kt, int buf) {
    const int kb = kt*32;
    {
      const unsigned lk = lds_off(&sk[buf][0]) + (unsigned)lane*144u;
      const uint64_t gk = (uint64_t)(k + ((size_t)(b*S_ + kb + lane))*D_ + h*HD_);
#pragma unroll
      for (int i=0;i<8;i++) async_b128(lk + i*16u, gk + i*16u);   // 64 bf16 = 8 x b128
    }
#pragma unroll
    for (int dd=lane; dd<64; dd+=32) {
      const unsigned lv = lds_off(&sv[buf][0]) + (unsigned)dd*80u;
      const uint64_t gv = (uint64_t)(vt + ((size_t)b*D_ + h*HD_ + dd)*S_ + kb);
#pragma unroll
      for (int i=0;i<4;i++) async_b128(lv + i*16u, gv + i*16u);   // 32 bf16 = 4 x b128
    }
  };

  FragBF aq[2];
  {
    const bf16* qr = q + ((size_t)(b*S_ + q0 + ll))*D_ + h*HD_;
#pragma unroll
    for (int c=0;c<2;c++) {
      aq[c].q[0] = *(const uint4*)(qr + c*32 + g*8);
      aq[c].q[1] = *(const uint4*)(qr + c*32 + 16 + g*8);
    }
  }
  v8f oacc[4];
  float mrow[8], lrow[8];
#pragma unroll
  for (int r=0;r<8;r++) { mrow[r] = -3.0e38f; lrow[r] = 0.f; }
#pragma unroll
  for (int dt=0;dt<4;dt++)
#pragma unroll
    for (int r=0;r<8;r++) oacc[dt][r] = 0.f;

  auto do_tile = [&](int kt) {
    const int kb = kt*32;
    const bf16* skc = &sk[kt&1][0];
    const bf16* svc = &sv[kt&1][0];
    v8f sc[2];
#pragma unroll
    for (int j=0;j<2;j++) {
      v8f a;
#pragma unroll
      for (int r=0;r<8;r++) a[r]=0.f;
#pragma unroll
      for (int c=0;c<2;c++) {
        FragBF bk;
        const bf16* base = skc + (j*16+ll)*72 + c*32 + g*8;
        bk.q[0] = *(const uint4*)base; bk.q[1] = *(const uint4*)(base+16);
        a = wmma_bf16(aq[c], bk, a);
      }
      sc[j] = a;
    }
#pragma unroll
    for (int r=0;r<8;r++) {
      const int rowi = q0 + r + 8*g;
      float s0 = sc[0][r]*0.125f; if (kb + ll      > rowi) s0 = -1.0e30f;
      float s1 = sc[1][r]*0.125f; if (kb + 16 + ll > rowi) s1 = -1.0e30f;
      float mx = fmaxf(s0, s1);
#pragma unroll
      for (int off=1; off<16; off<<=1) mx = fmaxf(mx, __shfl_xor(mx, off));
      const float mn = fmaxf(mrow[r], mx);
      const float alpha = __expf(mrow[r]-mn);
      const float e0 = __expf(s0-mn), e1 = __expf(s1-mn);
      float rs = e0 + e1;
#pragma unroll
      for (int off=1; off<16; off<<=1) rs += __shfl_xor(rs, off);
      mrow[r] = mn; lrow[r] = lrow[r]*alpha + rs;
#pragma unroll
      for (int dt=0;dt<4;dt++) oacc[dt][r] *= alpha;
      sp[(r+8*g)*40 + ll]      = (bf16)e0;
      sp[(r+8*g)*40 + 16 + ll] = (bf16)e1;
    }
    FragBF ap;
    ap.q[0] = *(const uint4*)(sp + ll*40 + g*8);
    ap.q[1] = *(const uint4*)(sp + ll*40 + 16 + g*8);
#pragma unroll
    for (int dt=0;dt<4;dt++) {
      FragBF bv;
      const bf16* vb = svc + (dt*16+ll)*40 + g*8;
      bv.q[0] = *(const uint4*)vb; bv.q[1] = *(const uint4*)(vb+16);
      oacc[dt] = wmma_bf16(ap, bv, oacc[dt]);
    }
  };

  const int ktn = q0/32 + 1;
  issue_kv(0, 0);
  for (int kt=0; kt<ktn-1; ++kt) {
    issue_kv(kt+1, (kt+1)&1);
    WAIT_ASYNC_HEX("0x10");
    do_tile(kt);
  }
  WAIT_ASYNC_HEX("0x0");
  do_tile(ktn-1);

#pragma unroll
  for (int dt=0;dt<4;dt++)
#pragma unroll
    for (int r=0;r<8;r++) {
      size_t idx = ((size_t)(b*S_ + q0 + r + 8*g))*D_ + h*HD_ + dt*16 + ll;
      o[idx] = (bf16)(oacc[dt][r] / lrow[r]);
    }
}

// ---------------------------------------------------------------- MoE routing
__global__ __launch_bounds__(256) void fill_kernel(int* row_tok, int* row_slot, float* row_gate, int* cnt) {
  int i = blockIdx.x*256 + threadIdx.x;
  if (i < MAXROWS_) { row_tok[i] = -1; row_slot[i] = -1; row_gate[i] = 0.f; }
  if (i < NE_) cnt[i] = 0;
}

__global__ __launch_bounds__(256) void gate_kernel(const bf16* __restrict__ h2, const float* __restrict__ gw,
    int* __restrict__ tok_exp, int* __restrict__ tok_pos, float* __restrict__ tok_gate, int* __restrict__ cnt) {
  const int t = blockIdx.x*8 + (threadIdx.x>>5);
  const int lane = threadIdx.x & 31;
  float acc[NE_];
#pragma unroll
  for (int e=0;e<NE_;e++) acc[e] = 0.f;
  const bf16* hr = h2 + (size_t)t*D_;
  for (int kk=lane; kk<D_; kk+=32) {
    float hv = (float)hr[kk];
    const float* gr = gw + (size_t)kk*NE_;
#pragma unroll
    for (int e=0;e<NE_;e++) acc[e] += hv*gr[e];
  }
#pragma unroll
  for (int e=0;e<NE_;e++)
#pragma unroll
    for (int off=16; off>=1; off>>=1) acc[e] += __shfl_xor(acc[e], off);
  if (lane==0) {
    int b0=0; float v0=acc[0];
    for (int e=1;e<NE_;e++) if (acc[e]>v0) { v0=acc[e]; b0=e; }
    int b1=-1; float v1=-3e38f;
    for (int e=0;e<NE_;e++) if (e!=b0 && acc[e]>v1) { v1=acc[e]; b1=e; }
    float mx = fmaxf(v0,v1);
    float e0 = __expf(v0-mx), e1 = __expf(v1-mx);
    float inv = 1.f/(e0+e1);
    int p0 = atomicAdd(&cnt[b0],1);
    int p1 = atomicAdd(&cnt[b1],1);
    tok_exp[2*t]=b0;   tok_pos[2*t]=p0;   tok_gate[2*t]=e0*inv;
    tok_exp[2*t+1]=b1; tok_pos[2*t+1]=p1; tok_gate[2*t+1]=e1*inv;
  }
}

__global__ void scan_kernel(const int* __restrict__ cnt, int* __restrict__ off, int* __restrict__ mblk) {
  if (threadIdx.x==0 && blockIdx.x==0) {
    int o = 0;
    for (int e=0;e<NE_;e++) {
      off[e] = o;
      int pc = ((cnt[e]+127)>>7)<<7;
      for (int mb=o>>7; mb<((o+pc)>>7); ++mb) mblk[mb] = e;
      o += pc;
    }
    off[NE_] = o;
    for (int mb=o>>7; mb<(MAXROWS_>>7); ++mb) mblk[mb] = 0;
  }
}

__global__ __launch_bounds__(256) void scatter_kernel(const int* tok_exp, const int* tok_pos, const float* tok_gate,
                               const int* off, int* row_tok, int* row_slot, float* row_gate) {
  int i = blockIdx.x*256 + threadIdx.x;
  if (i >= T_*TOPK_) return;
  int e = tok_exp[i];
  int r = off[e] + tok_pos[i];
  row_tok[r]  = i>>1;
  row_slot[r] = i;
  row_gate[r] = tok_gate[i];
}

__global__ __launch_bounds__(256) void combine_kernel(const float* __restrict__ x1, const float* __restrict__ y,
                                                      float* __restrict__ out) {
  size_t i = (size_t)blockIdx.x*256 + threadIdx.x;
  if (i >= (size_t)T_*D_) return;
  size_t t = i / D_, d = i % D_;
  out[i] = x1[i] + y[(size_t)(2*t)*D_ + d] + y[(size_t)(2*t+1)*D_ + d];
}

// ----------------------------------------------------------------
extern "C" void kernel_launch(void* const* d_in, const int* in_sizes, int n_in,
                              void* d_out, int out_size, void* d_ws, size_t ws_size,
                              hipStream_t stream) {
  (void)in_sizes; (void)n_in; (void)out_size; (void)ws_size;
  const float* x     = (const float*)d_in[0];
  const float* g1    = (const float*)d_in[1];
  const float* Wq    = (const float*)d_in[2];
  const float* Wdkv  = (const float*)d_in[3];
  const float* Wuk   = (const float*)d_in[4];
  const float* Wuv   = (const float*)d_in[5];
  const float* Wo    = (const float*)d_in[6];
  const float* g2    = (const float*)d_in[7];
  const float* gatew = (const float*)d_in[8];
  const float* w1    = (const float*)d_in[9];
  const float* w2    = (const float*)d_in[10];
  float* out = (float*)d_out;

  char* p = (char*)d_ws;
  auto alloc = [&](size_t bytes)->void* { void* r = (void*)p; p += (bytes+255)&~(size_t)255; return r; };

  bf16* h1    = (bf16*)alloc((size_t)T_*D_*2);
  bf16* wq_t  = (bf16*)alloc((size_t)D_*D_*2);
  bf16* wdkv_t= (bf16*)alloc((size_t)D_*LAT_*2);
  bf16* wuk_t = (bf16*)alloc((size_t)LAT_*D_*2);
  bf16* wuv_t = (bf16*)alloc((size_t)LAT_*D_*2);
  bf16* wo_t  = (bf16*)alloc((size_t)D_*D_*2);
  bf16* w1_t  = (bf16*)alloc((size_t)NE_*D_*ED_*2);
  bf16* w2_t  = (bf16*)alloc((size_t)NE_*ED_*D_*2);
  bf16* qb    = (bf16*)alloc((size_t)T_*D_*2);
  bf16* cb    = (bf16*)alloc((size_t)T_*LAT_*2);
  bf16* kbuf  = (bf16*)alloc((size_t)T_*D_*2);
  bf16* vtb   = (bf16*)alloc((size_t)T_*D_*2);
  bf16* attnb = (bf16*)alloc((size_t)T_*D_*2);
  float* x1   = (float*)alloc((size_t)T_*D_*4);
  bf16* h2b   = (bf16*)alloc((size_t)T_*D_*2);
  int*  tok_exp  = (int*)alloc((size_t)T_*TOPK_*4);
  int*  tok_pos  = (int*)alloc((size_t)T_*TOPK_*4);
  float* tok_gate= (float*)alloc((size_t)T_*TOPK_*4);
  int*  cnt      = (int*)alloc(NE_*4);
  int*  offs     = (int*)alloc((NE_+1)*4);
  int*  mblk     = (int*)alloc((MAXROWS_/128)*4);
  int*  row_tok  = (int*)alloc(MAXROWS_*4);
  int*  row_slot = (int*)alloc(MAXROWS_*4);
  float* row_gate= (float*)alloc(MAXROWS_*4);
  bf16* hidb  = (bf16*)alloc((size_t)MAXROWS_*ED_*2);
  float* yb   = (float*)alloc((size_t)T_*TOPK_*D_*4);

  const dim3 tb(32,8);
  cvt_t_kernel<<<dim3(D_/32,   D_/32, 1),   tb, 0, stream>>>(Wq,   wq_t,   D_,   D_);
  cvt_t_kernel<<<dim3(LAT_/32, D_/32, 1),   tb, 0, stream>>>(Wdkv, wdkv_t, D_,   LAT_);
  cvt_t_kernel<<<dim3(D_/32,   LAT_/32, 1), tb, 0, stream>>>(Wuk,  wuk_t,  LAT_, D_);
  cvt_t_kernel<<<dim3(D_/32,   LAT_/32, 1), tb, 0, stream>>>(Wuv,  wuv_t,  LAT_, D_);
  cvt_t_kernel<<<dim3(D_/32,   D_/32, 1),   tb, 0, stream>>>(Wo,   wo_t,   D_,   D_);
  cvt_t_kernel<<<dim3(ED_/32,  D_/32, NE_), tb, 0, stream>>>(w1,   w1_t,   D_,   ED_);
  cvt_t_kernel<<<dim3(D_/32,   ED_/32, NE_),tb, 0, stream>>>(w2,   w2_t,   ED_,  D_);

  fill_kernel<<<(MAXROWS_+255)/256, 256, 0, stream>>>(row_tok, row_slot, row_gate, cnt);

  rmsnorm_kernel<<<T_, 256, 0, stream>>>(x, g1, h1);

  gemm_kernel<<<dim3(D_/128,  T_/128), 256, 0, stream>>>(h1, wq_t,   T_, D_,   D_,   nullptr, nullptr, 0, 1, nullptr, qb,   nullptr, 0, nullptr, nullptr);
  gemm_kernel<<<dim3(LAT_/128,T_/128), 256, 0, stream>>>(h1, wdkv_t, T_, LAT_, D_,   nullptr, nullptr, 0, 1, nullptr, cb,   nullptr, 0, nullptr, nullptr);
  gemm_kernel<<<dim3(D_/128,  T_/128), 256, 0, stream>>>(cb, wuk_t,  T_, D_,   LAT_, nullptr, nullptr, 0, 1, nullptr, kbuf, nullptr, 0, nullptr, nullptr);
  gemm_kernel<<<dim3(D_/128,  T_/128), 256, 0, stream>>>(cb, wuv_t,  T_, D_,   LAT_, nullptr, nullptr, 0, 2, nullptr, vtb,  nullptr, S_, nullptr, nullptr);

  attn_kernel<<<dim3(S_/16, HEADS_, B_), 32, 0, stream>>>(qb, kbuf, vtb, attnb);

  gemm_kernel<<<dim3(D_/128, T_/128), 256, 0, stream>>>(attnb, wo_t, T_, D_, D_, nullptr, nullptr, 0, 0, x1, nullptr, x, 0, nullptr, nullptr);

  rmsnorm_kernel<<<T_, 256, 0, stream>>>(x1, g2, h2b);

  gate_kernel<<<T_/8, 256, 0, stream>>>(h2b, gatew, tok_exp, tok_pos, tok_gate, cnt);
  scan_kernel<<<1, 32, 0, stream>>>(cnt, offs, mblk);
  scatter_kernel<<<(T_*TOPK_+255)/256, 256, 0, stream>>>(tok_exp, tok_pos, tok_gate, offs, row_tok, row_slot, row_gate);

  gemm_kernel<<<dim3(ED_/128, MAXROWS_/128), 256, 0, stream>>>(h2b, w1_t, MAXROWS_, ED_, D_,
      row_tok, mblk, (long long)D_*ED_, 3, nullptr, hidb, nullptr, 0, nullptr, nullptr);
  gemm_kernel<<<dim3(D_/128, MAXROWS_/128), 256, 0, stream>>>(hidb, w2_t, MAXROWS_, D_, ED_,
      nullptr, mblk, (long long)ED_*D_, 4, yb, nullptr, nullptr, 0, row_slot, row_gate);

  combine_kernel<<<((size_t)T_*D_+255)/256, 256, 0, stream>>>(x1, yb, out);
}
